// MultilayerGRU_77627238908288
// MI455X (gfx1250) — matrix operand order repaired and checked by reading, not compiled, not run on hardware
//
#include <hip/hip_runtime.h>
#include <hip/hip_bf16.h>
#include <math.h>

// Problem constants (from reference)
#define B_ 64
#define S_ 1024
#define I_ 128
#define H_ 512
#define L_ 3
#define O_ 128

typedef __bf16 bf16_t;
typedef __attribute__((ext_vector_type(16))) __bf16 v16bf;
typedef __attribute__((ext_vector_type(8)))  float  v8f;
typedef unsigned int uint32x4v __attribute__((ext_vector_type(4)));
typedef int          int32x4v  __attribute__((ext_vector_type(4)));
typedef int          int32x8v  __attribute__((ext_vector_type(8)));

#define NSLICE 16                  // column-slices per (layer,gate); 32 cols each
#define GATE_WGS (L_ * 3 * NSLICE) // 144
#define OUT_WGS  (O_ / 16)         // 8
#define NWG      (GATE_WGS + OUT_WGS)
#define TPB      256

// dynamic LDS partition (bytes)
#define SWH_OFF 0u        // 32 KB: Wh (or Wy) slice
#define SWX_OFF 32768u    // 32 KB: Wx slice
#define SA_OFF  65536u    // 64 KB: TDM-staged activation panel (A matrix)
#define SMEM_BYTES 131072u

#if __has_include(<hip/amd_detail/amd_gfx1250_TDM.h>)
#define TDM_SIX_ARGS 1
#endif

// ---------------- workspace layout (bytes) ----------------
constexpr size_t OFF_INBF = 0;
constexpr size_t SZ_INBF  = (size_t)B_ * S_ * I_ * 2;
constexpr size_t OFF_WX0  = OFF_INBF + SZ_INBF;
constexpr size_t SZ_WX0   = (size_t)3 * H_ * I_ * 2;
constexpr size_t OFF_WX   = OFF_WX0 + SZ_WX0;
constexpr size_t SZ_WX    = (size_t)3 * (L_ - 1) * H_ * H_ * 2;
constexpr size_t OFF_WH   = OFF_WX + SZ_WX;
constexpr size_t SZ_WH    = (size_t)3 * L_ * H_ * H_ * 2;
constexpr size_t OFF_WY   = OFF_WH + SZ_WH;
constexpr size_t SZ_WY    = (size_t)O_ * H_ * 2;
constexpr size_t OFF_HBF  = OFF_WY + SZ_WY;          // bf16 h, [L][B][H]
constexpr size_t SZ_HBF   = (size_t)L_ * B_ * H_ * 2;
constexpr size_t OFF_RHB  = OFF_HBF + SZ_HBF;        // bf16 r*h, [L][B][H]
constexpr size_t SZ_RHB   = SZ_HBF;
constexpr size_t OFF_HF32 = OFF_RHB + SZ_RHB;        // f32 h, [L][B][H]
constexpr size_t SZ_HF32  = (size_t)L_ * B_ * H_ * 4;
constexpr size_t OFF_ZB   = OFF_HF32 + SZ_HF32;      // f32 z, [L][B][H]
constexpr size_t SZ_ZB    = SZ_HF32;
constexpr size_t OFF_BAR  = OFF_ZB + SZ_ZB;          // 2 x u32 barrier state

// ---------------- WMMA fragment helpers ----------------
union Frag { v16bf v; uint4 q[2]; };

// A (16x32 tile) fragment per ISA layout: lane L (r=L&15, h=L>>4) holds row
// m=mbase+r; elems 0..7 = K kbase+8h.., elems 8..15 = K kbase+16+8h..
__device__ __forceinline__ v16bf load_frag_a(const bf16_t* src, int ld,
                                             int mbase, int kbase, int lane) {
    int r = lane & 15, h = lane >> 4;
    const bf16_t* p = src + (size_t)(mbase + r) * ld + kbase + h * 8;
    Frag f;
    f.q[0] = *(const uint4*)(p);
    f.q[1] = *(const uint4*)(p + 16);
    return f.v;
}

// B (32x16 tile) of h @ W^T: B[k][n] = W[n][k]; lane L (n=L&15, h=L>>4) holds
// N=nbase+n, elems 0..7 = K kbase+8h.., elems 8..15 = K kbase+16+8h.. of row-major W[n][k]
__device__ __forceinline__ v16bf load_frag_b(const bf16_t* src, int ld,
                                             int nbase, int kbase, int lane) {
    int n = lane & 15, h = lane >> 4;
    const bf16_t* p = src + (size_t)(nbase + n) * ld + kbase + h * 8;
    Frag f;
    f.q[0] = *(const uint4*)(p);
    f.q[1] = *(const uint4*)(p + 16);
    return f.v;
}

__device__ __forceinline__ void gemm_acc(v8f& acc, const bf16_t* A, int lda,
                                         int mbase, const bf16_t* sB, int ldb, int nbase,
                                         int K, int lane) {
#pragma unroll 4
    for (int k = 0; k < K; k += 32) {
        v16bf a = load_frag_a(A, lda, mbase, k, lane);
        v16bf b = load_frag_b(sB, ldb, nbase, k, lane);
        acc = __builtin_amdgcn_wmma_f32_16x16x32_bf16(false, a, false, b,
                                                      (short)0, acc, false, false);
    }
}

// ---------------- Tensor Data Mover: 2D panel -> LDS ----------------
// D# per CDNA5 ISA ch.8: group0 = {flags, lds_addr, global_addr, type=2},
// group1 = {data_size=2B, tensor dims, tile dims, dim0 stride}.  One DMA per
// WG (issued by wave 0), tracked with TENSORcnt, consumed after syncthreads.
__device__ __forceinline__ void panel_issue(bf16_t* dstLds, unsigned lds_off,
                                            const bf16_t* src, unsigned rows,
                                            unsigned cols, unsigned stride_elems) {
#if __has_builtin(__builtin_amdgcn_tensor_load_to_lds)
    unsigned long long ga = (unsigned long long)(uintptr_t)src;
    uint32x4v g0;
    g0[0] = 1u;                                            // count=1, user D#
    g0[1] = lds_off;                                       // lds_addr (bytes)
    g0[2] = (unsigned)(ga & 0xffffffffu);                  // global_addr lo
    g0[3] = (unsigned)((ga >> 32) & 0x01ffffffu) | (2u << 30); // addr[56:32]|type=2
    int32x8v g1;
    g1[0] = (int)(1u << 16);                               // data_size: 2 bytes
    g1[1] = (int)((cols & 0xffffu) << 16);                 // tensor_dim0 lo16
    g1[2] = (int)(((cols >> 16) & 0xffffu) | ((rows & 0xffffu) << 16)); // dim0 hi | dim1 lo
    g1[3] = (int)(((rows >> 16) & 0xffffu) | ((cols & 0xffffu) << 16)); // dim1 hi | tile_dim0
    g1[4] = (int)(rows & 0xffffu);                         // tile_dim1 (tile_dim2=0)
    g1[5] = (int)stride_elems;                             // tensor_dim0_stride lo32
    g1[6] = 0;
    g1[7] = 0;
    int32x4v z4 = {0, 0, 0, 0};
#ifdef TDM_SIX_ARGS
    int32x8v z8 = {0, 0, 0, 0, 0, 0, 0, 0};
    __builtin_amdgcn_tensor_load_to_lds(g0, g1, z4, z4, z8, 0);
#else
    __builtin_amdgcn_tensor_load_to_lds(g0, g1, z4, z4, 0);
#endif
    (void)dstLds;
#else
    // fallback: wave-0 cooperative copy
    int l = threadIdx.x & 31;
    for (unsigned r = 0; r < rows; ++r)
        for (unsigned c = l * 8u; c < cols; c += 32u * 8u)
            *(uint4*)(dstLds + (size_t)r * cols + c) =
                *(const uint4*)(src + (size_t)r * stride_elems + c);
#endif
}

__device__ __forceinline__ void panel_wait() {
#if __has_builtin(__builtin_amdgcn_tensor_load_to_lds)
    __builtin_amdgcn_s_wait_tensorcnt(0);
#endif
}

// ---------------- grid-wide barrier (persistent kernel) ----------------
__device__ __forceinline__ void grid_sync(unsigned* cnt, unsigned* gen) {
    __syncthreads();
    if (threadIdx.x == 0) {
        __threadfence();
        unsigned g = __hip_atomic_load(gen, __ATOMIC_RELAXED, __HIP_MEMORY_SCOPE_AGENT);
        unsigned a = __hip_atomic_fetch_add(cnt, 1u, __ATOMIC_ACQ_REL, __HIP_MEMORY_SCOPE_AGENT);
        if (a == (unsigned)(NWG - 1)) {
            __hip_atomic_store(cnt, 0u, __ATOMIC_RELAXED, __HIP_MEMORY_SCOPE_AGENT);
            __hip_atomic_fetch_add(gen, 1u, __ATOMIC_ACQ_REL, __HIP_MEMORY_SCOPE_AGENT);
        } else {
            while (__hip_atomic_load(gen, __ATOMIC_ACQUIRE, __HIP_MEMORY_SCOPE_AGENT) == g)
                __builtin_amdgcn_s_sleep(2);
        }
        __threadfence();
    }
    __syncthreads();
}

// ---------------- prologue kernels ----------------
__global__ void cvt_bf16(const float* __restrict__ src, bf16_t* __restrict__ dst, size_t n) {
    size_t i = (size_t)blockIdx.x * blockDim.x + threadIdx.x;
    size_t stride = (size_t)gridDim.x * blockDim.x;
    for (; i < n; i += stride) dst[i] = (bf16_t)src[i];
}

__global__ void init_state(const float* __restrict__ hs, float* __restrict__ hf32,
                           bf16_t* __restrict__ hbf, unsigned* bar) {
    int i = blockIdx.x * blockDim.x + threadIdx.x;
    if (i == 0) { bar[0] = 0u; bar[1] = 0u; }
    int n = L_ * B_ * H_;
    int stride = gridDim.x * blockDim.x;
    for (; i < n; i += stride) {
        int l = i / (B_ * H_);
        int r = i % (B_ * H_);
        int b = r / H_;
        int h = r % H_;
        float v = hs[(size_t)b * L_ * H_ + (size_t)l * H_ + h];
        hf32[i] = v;
        hbf[i]  = (bf16_t)v;
    }
}

// ---------------- persistent wavefront GRU kernel ----------------
// wg < 144 : gate WG (layer = wg/48, gate = (wg%48)/16: 0=z,1=r,2=g, 32 cols)
// wg >= 144: output-projection WG (16 cols of O).
// Pipeline: layer i handles t = tick - i; output handles t = tick - 3.
__global__ __launch_bounds__(TPB)
void gru_persistent(const bf16_t* __restrict__ inbf,  // [B][S][I]
                    const bf16_t* __restrict__ wx0,   // [gate][H][I]
                    const bf16_t* __restrict__ wx,    // [gate][L-1][H][H]
                    const bf16_t* __restrict__ wh,    // [gate][L][H][H]
                    const bf16_t* __restrict__ wy,    // [O][H]
                    bf16_t* __restrict__ hbf, bf16_t* __restrict__ rhb,
                    float* __restrict__ hf32, float* __restrict__ zb,
                    const float* __restrict__ bz, const float* __restrict__ br,
                    const float* __restrict__ bg, const float* __restrict__ by,
                    float* __restrict__ out_y, float* __restrict__ out_h,
                    unsigned* bar_cnt, unsigned* bar_gen) {
    extern __shared__ __align__(16) char smem[];
    bf16_t* sWh = (bf16_t*)(smem + SWH_OFF);
    bf16_t* sWx = (bf16_t*)(smem + SWX_OFF);
    bf16_t* sA  = (bf16_t*)(smem + SA_OFF);

    const int wg   = blockIdx.x;
    const int tid  = threadIdx.x;
    const int lane = tid & 31;
    const int wave = tid >> 5;
    const int hi_  = lane >> 4;
    const int nl   = lane & 15;
    const bool isGate = wg < GATE_WGS;

    int layer = 0, gate = 0, colbase = 0;
    if (isGate) {
        layer = wg / (3 * NSLICE);
        int rem = wg % (3 * NSLICE);
        gate = rem / NSLICE;
        colbase = (rem % NSLICE) * 32;
        const bf16_t* whsrc = wh + ((size_t)gate * L_ + layer) * H_ * H_ + (size_t)colbase * H_;
        for (int i = tid; i < 32 * H_ / 8; i += TPB)
            ((uint4*)sWh)[i] = ((const uint4*)whsrc)[i];
        if (layer == 0) {
            const bf16_t* wxsrc = wx0 + (size_t)gate * H_ * I_ + (size_t)colbase * I_;
            for (int i = tid; i < 32 * I_ / 8; i += TPB)
                ((uint4*)sWx)[i] = ((const uint4*)wxsrc)[i];
        } else {
            const bf16_t* wxsrc = wx + ((size_t)gate * (L_ - 1) + (layer - 1)) * H_ * H_
                                     + (size_t)colbase * H_;
            for (int i = tid; i < 32 * H_ / 8; i += TPB)
                ((uint4*)sWx)[i] = ((const uint4*)wxsrc)[i];
        }
    } else {
        colbase = (wg - GATE_WGS) * 16;
        const bf16_t* wysrc = wy + (size_t)colbase * H_;
        for (int i = tid; i < 16 * H_ / 8; i += TPB)
            ((uint4*)sWh)[i] = ((const uint4*)wysrc)[i];
    }
    __syncthreads();

    // 8 waves: 4 M-tiles x 2 N-tiles of the 64x32 block
    const int mbase = (wave >> 1) * 16;
    const int nloc  = (wave & 1) * 16;

    for (int tick = 0; tick < S_ + 3; ++tick) {
        v8f gx = {}; // g-gate x-side partial, lives in VGPRs across the barrier
        const int t = isGate ? (tick - layer) : (tick - 3);
        const bool active = (t >= 0) && (t < S_);

        // ---- stage 1: TDM-stage the LDS A-panel; z/r overlap with x-GEMM ----
        if (active && wave == 0) {
            if (!isGate) {
                panel_issue(sA, SA_OFF, hbf + (size_t)2 * B_ * H_, B_, H_, H_);
            } else if (gate == 2) {
                if (layer == 0) panel_issue(sA, SA_OFF, inbf + (size_t)t * I_, B_, I_, S_ * I_);
                else            panel_issue(sA, SA_OFF, hbf + (size_t)(layer - 1) * B_ * H_, B_, H_, H_);
            } else {
                panel_issue(sA, SA_OFF, hbf + (size_t)layer * B_ * H_, B_, H_, H_);
            }
        }

        v8f acc = {};
        if (isGate && active && gate != 2) {
            // x-side GEMM straight from global (overlaps the TDM of the h-panel)
            if (layer == 0)
                gemm_acc(acc, inbf + (size_t)t * I_, S_ * I_, mbase, sWx, I_, nloc, I_, lane);
            else
                gemm_acc(acc, hbf + (size_t)(layer - 1) * B_ * H_, H_, mbase,
                         sWx, H_, nloc, H_, lane);
        }
        if (active && wave == 0) panel_wait();
        __syncthreads();

        if (isGate && active) {
            const int col = colbase + nloc + nl;
            if (gate == 2) {
                // g x-side from the TDM-staged panel
                if (layer == 0) gemm_acc(acc, sA, I_, mbase, sWx, I_, nloc, I_, lane);
                else            gemm_acc(acc, sA, H_, mbase, sWx, H_, nloc, H_, lane);
                float bias = bg[layer * H_ + col];
#pragma unroll
                for (int d = 0; d < 8; ++d) gx[d] = acc[d] + bias;
            } else {
                // + h_layer(t-1) @ Wh^T, A-panel from LDS
                gemm_acc(acc, sA, H_, mbase, sWh, H_, nloc, H_, lane);
                const float* bptr = (gate == 0) ? bz : br;
                float bias = bptr[layer * H_ + col];
#pragma unroll
                for (int d = 0; d < 8; ++d) {
                    int m = mbase + d + 8 * hi_;
                    float s = 1.0f / (1.0f + expf(-(acc[d] + bias)));
                    size_t idx = (size_t)layer * B_ * H_ + (size_t)m * H_ + col;
                    if (gate == 0) zb[idx] = s;                        // z (f32)
                    else           rhb[idx] = (bf16_t)(s * hf32[idx]); // r*h (bf16)
                }
            }
        } else if (!isGate && active && wave < 4) {
            // y(t) = h_2(t) @ Wy^T + by
            const int mb = wave * 16;
            const int col = colbase + nl;
            v8f yacc = {};
            gemm_acc(yacc, sA, H_, mb, sWh, H_, 0, H_, lane);
            float bias = by[col];
#pragma unroll
            for (int d = 0; d < 8; ++d) {
                int m = mb + d + 8 * hi_;
                out_y[((size_t)m * S_ + t) * O_ + col] = yacc[d] + bias;
            }
        }

        grid_sync(bar_cnt, bar_gen);

        // ---- stage 2: g-gate h-side (A = r*h, TDM-staged) + h update ----
        const bool doG = isGate && gate == 2 && active;
        if (doG && wave == 0) {
            panel_issue(sA, SA_OFF, rhb + (size_t)layer * B_ * H_, B_, H_, H_);
            panel_wait();
        }
        __syncthreads();
        if (doG) {
            const int col = colbase + nloc + nl;
            v8f acc2 = gx;
            gemm_acc(acc2, sA, H_, mbase, sWh, H_, nloc, H_, lane);
#pragma unroll
            for (int d = 0; d < 8; ++d) {
                int m = mbase + d + 8 * hi_;
                size_t idx = (size_t)layer * B_ * H_ + (size_t)m * H_ + col;
                float g = tanhf(acc2[d]);
                float z = zb[idx];
                float hn = z * hf32[idx] + (1.0f - z) * g;
                hf32[idx] = hn;
                hbf[idx]  = (bf16_t)hn;
                if (t == S_ - 1)
                    out_h[(size_t)m * L_ * H_ + (size_t)layer * H_ + col] = hn;
            }
        }

        grid_sync(bar_cnt, bar_gen);
    }
}

// ---------------- host launcher ----------------
extern "C" void kernel_launch(void* const* d_in, const int* in_sizes, int n_in,
                              void* d_out, int out_size, void* d_ws, size_t ws_size,
                              hipStream_t stream) {
    const float* input = (const float*)d_in[0];
    const float* hs    = (const float*)d_in[1];
    const float* Wx0z  = (const float*)d_in[2];
    const float* Wx0r  = (const float*)d_in[3];
    const float* Wx0g  = (const float*)d_in[4];
    const float* Wxz   = (const float*)d_in[5];
    const float* Wxr   = (const float*)d_in[6];
    const float* Wxg   = (const float*)d_in[7];
    const float* Whz   = (const float*)d_in[8];
    const float* Whr   = (const float*)d_in[9];
    const float* Whg   = (const float*)d_in[10];
    const float* bz    = (const float*)d_in[11];
    const float* br    = (const float*)d_in[12];
    const float* bg    = (const float*)d_in[13];
    const float* Wy    = (const float*)d_in[14];
    const float* by    = (const float*)d_in[15];

    char* ws = (char*)d_ws;
    bf16_t* inbf = (bf16_t*)(ws + OFF_INBF);
    bf16_t* wx0  = (bf16_t*)(ws + OFF_WX0);
    bf16_t* wx   = (bf16_t*)(ws + OFF_WX);
    bf16_t* wh   = (bf16_t*)(ws + OFF_WH);
    bf16_t* wy   = (bf16_t*)(ws + OFF_WY);
    bf16_t* hbf  = (bf16_t*)(ws + OFF_HBF);
    bf16_t* rhb  = (bf16_t*)(ws + OFF_RHB);
    float*  hf32 = (float*)(ws + OFF_HF32);
    float*  zb   = (float*)(ws + OFF_ZB);
    unsigned* bar = (unsigned*)(ws + OFF_BAR);

    float* out_y = (float*)d_out;                // (B,S,O)
    float* out_h = out_y + (size_t)B_ * S_ * O_; // (B,L,H)

    const size_t HH = (size_t)H_ * H_;
    cvt_bf16<<<2048, 256, 0, stream>>>(input, inbf, (size_t)B_ * S_ * I_);
    cvt_bf16<<<128, 256, 0, stream>>>(Wx0z, wx0 + 0 * (size_t)H_ * I_, (size_t)H_ * I_);
    cvt_bf16<<<128, 256, 0, stream>>>(Wx0r, wx0 + 1 * (size_t)H_ * I_, (size_t)H_ * I_);
    cvt_bf16<<<128, 256, 0, stream>>>(Wx0g, wx0 + 2 * (size_t)H_ * I_, (size_t)H_ * I_);
    cvt_bf16<<<512, 256, 0, stream>>>(Wxz, wx + 0 * (L_ - 1) * HH, (L_ - 1) * HH);
    cvt_bf16<<<512, 256, 0, stream>>>(Wxr, wx + 1 * (L_ - 1) * HH, (L_ - 1) * HH);
    cvt_bf16<<<512, 256, 0, stream>>>(Wxg, wx + 2 * (L_ - 1) * HH, (L_ - 1) * HH);
    cvt_bf16<<<512, 256, 0, stream>>>(Whz, wh + 0 * (size_t)L_ * HH, (size_t)L_ * HH);
    cvt_bf16<<<512, 256, 0, stream>>>(Whr, wh + 1 * (size_t)L_ * HH, (size_t)L_ * HH);
    cvt_bf16<<<512, 256, 0, stream>>>(Whg, wh + 2 * (size_t)L_ * HH, (size_t)L_ * HH);
    cvt_bf16<<<128, 256, 0, stream>>>(Wy, wy, (size_t)O_ * H_);
    init_state<<<256, 256, 0, stream>>>(hs, hf32, hbf, bar);

    (void)hipFuncSetAttribute((const void*)gru_persistent,
                              hipFuncAttributeMaxDynamicSharedMemorySize,
                              (int)SMEM_BYTES);
    gru_persistent<<<NWG, TPB, SMEM_BYTES, stream>>>(inbf, wx0, wx, wh, wy,
                                                     hbf, rhb, hf32, zb,
                                                     bz, br, bg, by,
                                                     out_y, out_h, bar, bar + 1);
}
